// StandardAttention_47699906789402
// MI455X (gfx1250) — compile-verified, matrix operand order checked by reading, not run on hardware
//
#include <hip/hip_runtime.h>
#include <hip/hip_bf16.h>

typedef __bf16 bf16;
typedef __attribute__((ext_vector_type(16))) __bf16 v16bf;
typedef __attribute__((ext_vector_type(8)))  __bf16 bf16x8;
typedef __attribute__((ext_vector_type(8)))  float  v8f;
typedef __attribute__((ext_vector_type(4)))  int    v4i;

typedef __attribute__((address_space(1))) v4i gv4i;  // global int4
typedef __attribute__((address_space(3))) v4i lv4i;  // LDS int4

#if defined(__gfx1250__) && \
    __has_builtin(__builtin_amdgcn_global_load_async_to_lds_b128) && \
    __has_builtin(__builtin_amdgcn_s_wait_asynccnt)
#define USE_ASYNC_LDS 1
#else
#define USE_ASYNC_LDS 0
#endif

static constexpr int B_ = 2;
static constexpr int T_ = 2048;
static constexpr int C_ = 1024;
static constexpr int H_ = 16;
static constexpr int D_ = 64;       // head dim
static constexpr float SCALE_ = 0.125f;  // 1/sqrt(64)

// ---------------------------------------------------------------------------
// Fragment loaders (CDNA5 wave32 WMMA layouts, 16-bit data, 16x16x32)
// A-frag: lane&15 = M row; halves 0..7 -> K = kb+0..7, halves 8..15 -> K = kb+16..23
//         where kb = (lane>=16) ? 8 : 0.
// B-frag mirrors with N in the lane dimension; source stored "N-major" [N][K].
// Works for both global and LDS source pointers.
// ---------------------------------------------------------------------------
__device__ __forceinline__ v16bf load_frag(const bf16* __restrict__ base, int ld,
                                           int row0, int k0) {
  const int lane = threadIdx.x & 31;
  const int r    = lane & 15;
  const int kb   = (lane & 16) ? 8 : 0;
  const bf16* p  = base + (size_t)(row0 + r) * (size_t)ld + (k0 + kb);
  bf16x8 lo = *reinterpret_cast<const bf16x8*>(p);
  bf16x8 hi = *reinterpret_cast<const bf16x8*>(p + 16);
  v16bf f;
#pragma unroll
  for (int i = 0; i < 8; ++i) { f[i] = lo[i]; f[8 + i] = hi[i]; }
  return f;
}

__device__ __forceinline__ v8f wmma_bf16(v16bf a, v16bf b, v8f c) {
  return __builtin_amdgcn_wmma_f32_16x16x32_bf16(false, a, false, b,
                                                 (short)0, c, false, false);
}

// Reductions across the 16-lane half of a wave32 (rows of the C/D layout).
__device__ __forceinline__ float hmax16(float v) {
#pragma unroll
  for (int m = 1; m <= 8; m <<= 1) v = fmaxf(v, __shfl_xor(v, m, 32));
  return v;
}
__device__ __forceinline__ float hsum16(float v) {
#pragma unroll
  for (int m = 1; m <= 8; m <<= 1) v += __shfl_xor(v, m, 32);
  return v;
}

// ---------------------------------------------------------------------------
// Prep kernels: fp32 -> bf16 copies / transposes
// ---------------------------------------------------------------------------
__global__ void __launch_bounds__(256) cvt_bf16_kernel(const float* __restrict__ in,
                                                       bf16* __restrict__ out, int n) {
  int i = blockIdx.x * blockDim.x + threadIdx.x;
  if (i < n) out[i] = (bf16)in[i];
}

// out[n*K + k] = in[k*N + n]   (W[K][N] row-major -> Wt[N][K] bf16)
__global__ void __launch_bounds__(256) transpose_cvt_kernel(const float* __restrict__ in,
                                                            bf16* __restrict__ out,
                                                            int K, int N) {
  int i = blockIdx.x * blockDim.x + threadIdx.x;
  if (i < K * N) {
    int n = i / K, k = i - n * K;
    out[i] = (bf16)in[(size_t)k * N + n];
  }
}

// ---------------------------------------------------------------------------
// QKV GEMM: qkv = x_bf16 @ w_attn + b_attn, scattered into Q[B,H,T,D],
// K[B,H,T,D], Vt[B,H,D,T] (V transposed for contiguous PV B-fragments).
// One 16x64 tile per wave; software-pipelined K loop.
// ---------------------------------------------------------------------------
__global__ void __launch_bounds__(256) qkv_gemm_kernel(
    const bf16* __restrict__ xb, const bf16* __restrict__ wt,
    const float* __restrict__ bias,
    bf16* __restrict__ Q, bf16* __restrict__ K, bf16* __restrict__ Vt) {
  const int NT = (3 * C_) / 64;  // 48 column-groups of 64
  int wave = (blockIdx.x * blockDim.x + threadIdx.x) >> 5;
  int wm = wave / NT, wn = wave - wm * NT;
  int m0 = wm * 16, n0 = wn * 64;

  v8f acc[4];
#pragma unroll
  for (int j = 0; j < 4; ++j)
#pragma unroll
    for (int i = 0; i < 8; ++i) acc[j][i] = 0.0f;

  v16bf a  = load_frag(xb, C_, m0, 0);
  v16bf b0 = load_frag(wt, C_, n0 +  0, 0);
  v16bf b1 = load_frag(wt, C_, n0 + 16, 0);
  v16bf b2 = load_frag(wt, C_, n0 + 32, 0);
  v16bf b3 = load_frag(wt, C_, n0 + 48, 0);

#pragma unroll 2
  for (int k = 32; k < C_; k += 32) {
    v16bf an = load_frag(xb, C_, m0, k);
    v16bf c0 = load_frag(wt, C_, n0 +  0, k);
    v16bf c1 = load_frag(wt, C_, n0 + 16, k);
    v16bf c2 = load_frag(wt, C_, n0 + 32, k);
    v16bf c3 = load_frag(wt, C_, n0 + 48, k);

    acc[0] = wmma_bf16(a, b0, acc[0]);
    acc[1] = wmma_bf16(a, b1, acc[1]);
    acc[2] = wmma_bf16(a, b2, acc[2]);
    acc[3] = wmma_bf16(a, b3, acc[3]);

    a = an; b0 = c0; b1 = c1; b2 = c2; b3 = c3;
  }
  acc[0] = wmma_bf16(a, b0, acc[0]);
  acc[1] = wmma_bf16(a, b1, acc[1]);
  acc[2] = wmma_bf16(a, b2, acc[2]);
  acc[3] = wmma_bf16(a, b3, acc[3]);

  const int lane = threadIdx.x & 31;
  const int col  = lane & 15;
  const int hi8  = (lane & 16) ? 8 : 0;
  const int bidx = m0 / T_;
  const int t0   = m0 - bidx * T_;  // tiles never straddle the batch dim

#pragma unroll
  for (int j = 0; j < 4; ++j) {
    const int n  = n0 + 16 * j + col;
    const float bv = bias[n];
#pragma unroll
    for (int r = 0; r < 8; ++r) {
      int t = t0 + r + hi8;
      bf16 v = (bf16)(acc[j][r] + bv);
      if (n < C_) {
        int h = n >> 6, d = n & 63;
        Q[(((size_t)bidx * H_ + h) * T_ + t) * D_ + d] = v;
      } else if (n < 2 * C_) {
        int n2 = n - C_, h = n2 >> 6, d = n2 & 63;
        K[(((size_t)bidx * H_ + h) * T_ + t) * D_ + d] = v;
      } else {
        int n2 = n - 2 * C_, h = n2 >> 6, d = n2 & 63;
        Vt[(((size_t)bidx * H_ + h) * D_ + d) * T_ + t] = v;
      }
    }
  }
}

// ---------------------------------------------------------------------------
// Flash attention: one wave handles one (b,h, 16-row q-tile).
// K/V tiles are double-buffered in LDS via GLOBAL_LOAD_ASYNC_TO_LDS_B128
// (ASYNCcnt-tracked); the next block's copies overlap the current block's
// WMMAs, softmax, and LDS P round-trip. Falls back to direct global
// fragment loads if the async builtins are unavailable.
// ---------------------------------------------------------------------------
__global__ void __launch_bounds__(256) flash_attn_kernel(
    const bf16* __restrict__ Q, const bf16* __restrict__ K,
    const bf16* __restrict__ Vt, bf16* __restrict__ Y) {
  __shared__ __align__(16) bf16 lds_p[8][16 * 32];  // per-wave 16x32 P tile
#if USE_ASYNC_LDS
  // per-wave double buffer: [buf][0..2047]=K tile 32x64, [2048..4095]=V tile 64x32
  __shared__ __align__(16) bf16 kv_lds[8][2][4096];
#endif

  const int wave_in_blk = threadIdx.x >> 5;
  const int wave = (blockIdx.x * blockDim.x + threadIdx.x) >> 5;
  const int QT = T_ / 16;             // 128 q-tiles per head
  const int qt = wave % QT;
  const int bh = wave / QT;
  if (bh >= B_ * H_) return;

  const bf16* Qh = Q  + (size_t)bh * T_ * D_;
  const bf16* Kh = K  + (size_t)bh * T_ * D_;
  const bf16* Vh = Vt + (size_t)bh * D_ * T_;

  const int q0   = qt * 16;
  const int lane = threadIdx.x & 31;
  const int col  = lane & 15;
  const int hi8  = (lane & 16) ? 8 : 0;
  bf16* pbuf = &lds_p[wave_in_blk][0];

  // Q fragments for this tile (K-dim = 64 -> two 32-wide chunks)
  v16bf qa0 = load_frag(Qh, D_, q0, 0);
  v16bf qa1 = load_frag(Qh, D_, q0, 32);

  v8f o0, o1, o2, o3;
  float mrow[8], lrow[8];
#pragma unroll
  for (int i = 0; i < 8; ++i) {
    o0[i] = o1[i] = o2[i] = o3[i] = 0.0f;
    mrow[i] = -3.0e38f;
    lrow[i] = 0.0f;
  }

  const int nkb = qt / 2 + 1;  // causal: keys <= q0+15 -> (q0+15)/32 + 1 blocks

#if USE_ASYNC_LDS
  // Issue one 32-key block (K tile 4KB + V tile 4KB) = 16 async b128 ops.
  auto issue_async = [&](int kt0, int buf) {
    bf16* kdst = &kv_lds[wave_in_blk][buf][0];
    bf16* vdst = &kv_lds[wave_in_blk][buf][2048];
    const bf16* ksrc = Kh + (size_t)kt0 * D_;  // 32 rows x 64, contiguous
#pragma unroll
    for (int i = 0; i < 8; ++i) {
      int c = i * 32 + lane;  // 16-byte chunk index
      __builtin_amdgcn_global_load_async_to_lds_b128(
          (gv4i*)(ksrc + c * 8), (lv4i*)(kdst + c * 8), 0, 0);
    }
#pragma unroll
    for (int i = 0; i < 8; ++i) {
      int c = i * 32 + lane;
      int d = c >> 2, part = c & 3;  // V rows are d (64), 32 keys = 4 chunks/row
      __builtin_amdgcn_global_load_async_to_lds_b128(
          (gv4i*)(Vh + (size_t)d * T_ + kt0 + part * 8),
          (lv4i*)(vdst + c * 8), 0, 0);
    }
  };
  issue_async(0, 0);
#endif

  for (int kb = 0; kb < nkb; ++kb) {
    const int kt0 = kb * 32;

#if USE_ASYNC_LDS
    const int cur = kb & 1;
    if (kb + 1 < nkb) {
      issue_async(kt0 + 32, cur ^ 1);
      __builtin_amdgcn_s_wait_asynccnt(16);  // block kb's 16 ops are done
    } else {
      __builtin_amdgcn_s_wait_asynccnt(0);
    }
    const bf16* kbuf = &kv_lds[wave_in_blk][cur][0];
    const bf16* vbuf = &kv_lds[wave_in_blk][cur][2048];
    v16bf k00 = load_frag(kbuf, D_, 0,  0);
    v16bf k01 = load_frag(kbuf, D_, 0,  32);
    v16bf k10 = load_frag(kbuf, D_, 16, 0);
    v16bf k11 = load_frag(kbuf, D_, 16, 32);
    v16bf vb0 = load_frag(vbuf, 32, 0,  0);
    v16bf vb1 = load_frag(vbuf, 32, 16, 0);
    v16bf vb2 = load_frag(vbuf, 32, 32, 0);
    v16bf vb3 = load_frag(vbuf, 32, 48, 0);
#else
    if (kb + 1 < nkb) {
      __builtin_prefetch(Kh + (size_t)(kt0 + 32) * D_, 0, 1);
      __builtin_prefetch(Vh + (kt0 + 32), 0, 1);
    }
    v16bf k00 = load_frag(Kh, D_, kt0,      0);
    v16bf k01 = load_frag(Kh, D_, kt0,      32);
    v16bf k10 = load_frag(Kh, D_, kt0 + 16, 0);
    v16bf k11 = load_frag(Kh, D_, kt0 + 16, 32);
    v16bf vb0 = load_frag(Vh, T_, 0,  kt0);
    v16bf vb1 = load_frag(Vh, T_, 16, kt0);
    v16bf vb2 = load_frag(Vh, T_, 32, kt0);
    v16bf vb3 = load_frag(Vh, T_, 48, kt0);
#endif

    // S = Q @ K^T for 32 keys -> two 16x16 f32 tiles
    v8f s0, s1;
#pragma unroll
    for (int i = 0; i < 8; ++i) { s0[i] = 0.0f; s1[i] = 0.0f; }
    s0 = wmma_bf16(qa0, k00, s0);
    s0 = wmma_bf16(qa1, k01, s0);
    s1 = wmma_bf16(qa0, k10, s1);
    s1 = wmma_bf16(qa1, k11, s1);

    const bool edge = (kb == nkb - 1);
    const int key0 = kt0 + col, key1 = key0 + 16;

#pragma unroll
    for (int r = 0; r < 8; ++r) {
      const int qrow = q0 + r + hi8;
      float e0 = s0[r] * SCALE_;
      float e1 = s1[r] * SCALE_;
      if (edge) {
        if (key0 > qrow) e0 = -3.0e38f;
        if (key1 > qrow) e1 = -3.0e38f;
      }
      float mv   = hmax16(fmaxf(e0, e1));
      float mnew = fmaxf(mrow[r], mv);
      float alpha = __expf(mrow[r] - mnew);
      mrow[r] = mnew;
      float p0 = __expf(e0 - mnew);
      float p1 = __expf(e1 - mnew);
      lrow[r] = lrow[r] * alpha + hsum16(p0 + p1);
      o0[r] *= alpha; o1[r] *= alpha; o2[r] *= alpha; o3[r] *= alpha;
      const int prow = r + hi8;
      pbuf[prow * 32 + col]      = (bf16)p0;
      pbuf[prow * 32 + 16 + col] = (bf16)p1;
    }

    // Wave-private LDS round-trip: ensure stores landed before fragment reads.
    asm volatile("s_wait_dscnt 0" ::: "memory");

    // Re-load P as an A-fragment (lane = q row, halves = key offsets)
    v16bf pa;
    {
      const int prow = lane & 15;
      const int kbo  = (lane & 16) ? 8 : 0;
      const bf16* pp = pbuf + prow * 32 + kbo;
      bf16x8 lo = *reinterpret_cast<const bf16x8*>(pp);
      bf16x8 hi = *reinterpret_cast<const bf16x8*>(pp + 16);
#pragma unroll
      for (int i = 0; i < 8; ++i) { pa[i] = lo[i]; pa[8 + i] = hi[i]; }
    }

    // O += P @ V
    o0 = wmma_bf16(pa, vb0, o0);
    o1 = wmma_bf16(pa, vb1, o1);
    o2 = wmma_bf16(pa, vb2, o2);
    o3 = wmma_bf16(pa, vb3, o3);
  }

  // Epilogue: divide by l, write Y[b, t, h*64 + d] as bf16
  const int bidx = bh / H_;
  const int h    = bh - bidx * H_;
  const size_t ybase = (size_t)bidx * T_ * C_;
#pragma unroll
  for (int r = 0; r < 8; ++r) {
    const int t = q0 + r + hi8;
    const float inv = 1.0f / lrow[r];
    size_t off = ybase + (size_t)t * C_ + h * D_ + col;
    Y[off + 0]  = (bf16)(o0[r] * inv);
    Y[off + 16] = (bf16)(o1[r] * inv);
    Y[off + 32] = (bf16)(o2[r] * inv);
    Y[off + 48] = (bf16)(o3[r] * inv);
  }
}

// ---------------------------------------------------------------------------
// Output projection: out = Y_bf16 @ w_proj + b_proj (fp32 output)
// ---------------------------------------------------------------------------
__global__ void __launch_bounds__(256) proj_gemm_kernel(
    const bf16* __restrict__ Yb, const bf16* __restrict__ Wt,
    const float* __restrict__ bias, float* __restrict__ out) {
  const int NT = C_ / 64;  // 16 column-groups of 64
  int wave = (blockIdx.x * blockDim.x + threadIdx.x) >> 5;
  int wm = wave / NT, wn = wave - wm * NT;
  int m0 = wm * 16, n0 = wn * 64;

  v8f acc[4];
#pragma unroll
  for (int j = 0; j < 4; ++j)
#pragma unroll
    for (int i = 0; i < 8; ++i) acc[j][i] = 0.0f;

  v16bf a  = load_frag(Yb, C_, m0, 0);
  v16bf b0 = load_frag(Wt, C_, n0 +  0, 0);
  v16bf b1 = load_frag(Wt, C_, n0 + 16, 0);
  v16bf b2 = load_frag(Wt, C_, n0 + 32, 0);
  v16bf b3 = load_frag(Wt, C_, n0 + 48, 0);

#pragma unroll 2
  for (int k = 32; k < C_; k += 32) {
    v16bf an = load_frag(Yb, C_, m0, k);
    v16bf c0 = load_frag(Wt, C_, n0 +  0, k);
    v16bf c1 = load_frag(Wt, C_, n0 + 16, k);
    v16bf c2 = load_frag(Wt, C_, n0 + 32, k);
    v16bf c3 = load_frag(Wt, C_, n0 + 48, k);

    acc[0] = wmma_bf16(a, b0, acc[0]);
    acc[1] = wmma_bf16(a, b1, acc[1]);
    acc[2] = wmma_bf16(a, b2, acc[2]);
    acc[3] = wmma_bf16(a, b3, acc[3]);

    a = an; b0 = c0; b1 = c1; b2 = c2; b3 = c3;
  }
  acc[0] = wmma_bf16(a, b0, acc[0]);
  acc[1] = wmma_bf16(a, b1, acc[1]);
  acc[2] = wmma_bf16(a, b2, acc[2]);
  acc[3] = wmma_bf16(a, b3, acc[3]);

  const int lane = threadIdx.x & 31;
  const int col  = lane & 15;
  const int hi8  = (lane & 16) ? 8 : 0;
#pragma unroll
  for (int j = 0; j < 4; ++j) {
    const int n = n0 + 16 * j + col;
    const float bv = bias[n];
#pragma unroll
    for (int r = 0; r < 8; ++r) {
      out[(size_t)(m0 + r + hi8) * C_ + n] = acc[j][r] + bv;
    }
  }
}

// ---------------------------------------------------------------------------
// Host launcher
// ---------------------------------------------------------------------------
extern "C" void kernel_launch(void* const* d_in, const int* in_sizes, int n_in,
                              void* d_out, int out_size, void* d_ws, size_t ws_size,
                              hipStream_t stream) {
  const float* x      = (const float*)d_in[0];
  const float* w_attn = (const float*)d_in[1];
  const float* b_attn = (const float*)d_in[2];
  const float* w_proj = (const float*)d_in[3];
  const float* b_proj = (const float*)d_in[4];
  float* out = (float*)d_out;

  const size_t XN  = (size_t)B_ * T_ * C_;       // 4,194,304
  const size_t WAN = (size_t)C_ * 3 * C_;        // 3,145,728
  const size_t WPN = (size_t)C_ * C_;            // 1,048,576

  char* ws = (char*)d_ws;
  size_t off = 0;
  auto alloc = [&](size_t bytes) {
    void* p = ws + off;
    off += (bytes + 255) & ~(size_t)255;
    return p;
  };
  bf16* xb      = (bf16*)alloc(XN * sizeof(bf16));
  bf16* wt_attn = (bf16*)alloc(WAN * sizeof(bf16));
  bf16* wt_proj = (bf16*)alloc(WPN * sizeof(bf16));
  bf16* Q       = (bf16*)alloc(XN * sizeof(bf16));
  bf16* K       = (bf16*)alloc(XN * sizeof(bf16));
  bf16* Vt      = (bf16*)alloc(XN * sizeof(bf16));
  bf16* Y       = (bf16*)alloc(XN * sizeof(bf16));
  (void)ws_size; (void)in_sizes; (void)n_in; (void)out_size;

  const int TPB = 256;
  cvt_bf16_kernel<<<(int)((XN + TPB - 1) / TPB), TPB, 0, stream>>>(x, xb, (int)XN);
  transpose_cvt_kernel<<<(int)((WAN + TPB - 1) / TPB), TPB, 0, stream>>>(w_attn, wt_attn, C_, 3 * C_);
  transpose_cvt_kernel<<<(int)((WPN + TPB - 1) / TPB), TPB, 0, stream>>>(w_proj, wt_proj, C_, C_);

  // QKV: (4096/16) * (3072/64) = 12288 wave-tiles, 8 waves per 256-thread block
  qkv_gemm_kernel<<<12288 / 8, TPB, 0, stream>>>(xb, wt_attn, b_attn, Q, K, Vt);

  // Flash attention: B*H*(T/16) = 4096 wave-tiles
  flash_attn_kernel<<<4096 / 8, TPB, 0, stream>>>(Q, K, Vt, Y);

  // Projection: (4096/16) * (1024/64) = 4096 wave-tiles
  proj_gemm_kernel<<<4096 / 8, TPB, 0, stream>>>(Y, wt_proj, b_proj, out);
}
